// NetAttention_61426622267525
// MI455X (gfx1250) — matrix operand order repaired and checked by reading, not compiled, not run on hardware
//
#include <hip/hip_runtime.h>
#include <hip/hip_bf16.h>

// ---------------------------------------------------------------------------
// MI455X (gfx1250) fused multi-head neighbor attention.
// All GEMMs run on v_wmma_f32_16x16x32_bf16 (wave32, 16x16 tiles, K=32).
// ---------------------------------------------------------------------------

typedef __bf16 bf16;
typedef __attribute__((ext_vector_type(16))) __bf16 v16bf;
typedef __attribute__((ext_vector_type(8)))  __bf16 v8bf;
typedef __attribute__((ext_vector_type(4)))  __bf16 v4bf;
typedef __attribute__((ext_vector_type(8)))  float  v8f;

#define WMMA_BF16(A, B, C) \
    __builtin_amdgcn_wmma_f32_16x16x32_bf16(false, (A), false, (B), (short)0, (C), false, false)

#define BATCH 64
#define TIME  64
#define HID   128
#define NHEAD 8
#define NCHK  17      /* (MAX_NEIGH+1) chunks of 64 kv rows */
#define SEQ   1088    /* 17 * 64 */

// ---- fragment loaders (layouts per CDNA5 ISA 7.12.2) ----------------------
// A matrix 16x32 bf16: lanes 0-15 rows M=0..15 with K=[k0..k0+7]+[k0+16..k0+23],
// lanes 16-31 same rows with K shifted by +8.  Source is row-major (ldk elems).
__device__ __forceinline__ v16bf load_a_frag(const bf16* base, int row0, int k0,
                                             int ldk, int lane) {
    int r = lane & 15, hi = lane >> 4;
    const bf16* p = base + (size_t)(row0 + r) * ldk + k0 + hi * 8;
    v8bf lo = *(const v8bf*)(p);
    v8bf hh = *(const v8bf*)(p + 16);
    return __builtin_shufflevector(lo, hh, 0, 1, 2, 3, 4, 5, 6, 7,
                                   8, 9, 10, 11, 12, 13, 14, 15);
}

// B matrix 32x16 bf16: lane n = column, lanes 0-15 hold K=[k0..k0+7]+[k0+16..k0+23],
// lanes 16-31 same with K shifted by +8 (matches A).  Source stored "n-major"
// (row = n, contiguous along K with leading dim ldk).
__device__ __forceinline__ v16bf load_b_frag(const bf16* base, int n0, int k0,
                                             int ldk, int lane) {
    int n = lane & 15, hi = lane >> 4;
    const bf16* p = base + (size_t)(n0 + n) * ldk + k0 + hi * 8;
    v8bf lo = *(const v8bf*)(p);
    v8bf hh = *(const v8bf*)(p + 16);
    return __builtin_shufflevector(lo, hh, 0, 1, 2, 3, 4, 5, 6, 7,
                                   8, 9, 10, 11, 12, 13, 14, 15);
}

// stage a 64x128 f32 tile into LDS as bf16 (row-major), 128 threads
__device__ __forceinline__ void stage_chunk(bf16* dst, const float* __restrict__ src,
                                            int tid) {
    const float4* s4 = (const float4*)src;
#pragma unroll 4
    for (int i = tid; i < 2048; i += 128) {
        float4 v = s4[i];
        v4bf w = { (bf16)v.x, (bf16)v.y, (bf16)v.z, (bf16)v.w };
        *(v4bf*)(dst + i * 4) = w;
    }
}

// ---------------------------------------------------------------------------
// prep: bf16-transpose the per-head weights (so WMMA B-frags load contiguous)
// wqT/wkT/wvT layout: [h][e][d]  (n-major, ldk=128).  prjB: bf16 copy of
// prj_w row-major [e'][1024] (already n-major for the output projection).
// ---------------------------------------------------------------------------
__global__ void prep_weights(const float* __restrict__ wq, const float* __restrict__ wk,
                             const float* __restrict__ wv, const float* __restrict__ pw,
                             bf16* __restrict__ wqT, bf16* __restrict__ wkT,
                             bf16* __restrict__ wvT, bf16* __restrict__ prjB) {
    int i = blockIdx.x * 256 + threadIdx.x;
    if (i >= NHEAD * HID * HID) return;
    int h = i >> 14, rem = i & 16383, d = rem >> 7, e = rem & 127;
    int t = h * 16384 + e * 128 + d;
    wqT[t] = (bf16)wq[i];
    wkT[t] = (bf16)wk[i];
    wvT[t] = (bf16)wv[i];
    prjB[i] = (bf16)pw[i];   // 128*1024 == 8*128*128 elements
}

// seed out with the bias, zero the Asum region (heads accumulate atomically)
__global__ void init_out(float* __restrict__ out, const float* __restrict__ prj_b) {
    int i = blockIdx.x * 256 + threadIdx.x;
    if (i < BATCH * TIME * HID) out[i] = prj_b[i & 127];
    if (i < BATCH * TIME * SEQ) out[BATCH * TIME * HID + i] = 0.f;
}

// ---------------------------------------------------------------------------
// fused attention: one workgroup per (b, h); 4 waves, each owns 16 query rows
// ---------------------------------------------------------------------------
__global__ __launch_bounds__(128)
void attn_kernel(const float* __restrict__ node, const float* __restrict__ neigh,
                 const unsigned char* __restrict__ att_mask,
                 const bf16* __restrict__ wqT, const bf16* __restrict__ wkT,
                 const bf16* __restrict__ wvT, const bf16* __restrict__ prjB,
                 float* __restrict__ out, float* __restrict__ Asum) {
    __shared__ bf16 lds_chunk[TIME * HID];        // 16 KB: staged kv chunk (bf16)
    __shared__ bf16 lds_kc[TIME * HID];           // 16 KB: Kc row-major / Q relayout
    __shared__ bf16 lds_vt[HID * TIME];           // 16 KB: Vc transposed [e][s']
    __shared__ bf16 lds_p[4][16 * 64];            //  8 KB: per-wave P=exp(S-m)
    __shared__ unsigned char lds_mask[TIME * TIME]; // 4 KB

    const int b = blockIdx.x, h = blockIdx.y;
    const int tid = threadIdx.x;
    const int wave = tid >> 5, lane = tid & 31;
    const int r16 = lane & 15, hi = lane >> 4;
    const int m0 = wave * 16;                      // this wave's query-row tile

    // ---- stage node tile (= kv chunk 0) and the mask ----
    stage_chunk(lds_chunk, node + (size_t)b * (TIME * HID), tid);
    {
        const uint4* ms = (const uint4*)(att_mask + (size_t)b * (TIME * TIME));
        uint4* md = (uint4*)lds_mask;
#pragma unroll
        for (int i = tid; i < 256; i += 128) md[i] = ms[i];
    }
    __syncthreads();

    // mask bits repeat with period 64 along s -> hoist into a register bitmask
    unsigned mbits = 0;
#pragma unroll
    for (int nt = 0; nt < 4; ++nt)
#pragma unroll
        for (int r = 0; r < 8; ++r) {
            int t = m0 + r + 8 * hi;
            if (lds_mask[t * 64 + nt * 16 + r16]) mbits |= 1u << (nt * 8 + r);
        }

    // ---- Q projection: Q = node_tile @ Wq ; re-layout through LDS into A-frags
    v16bf afr[4];
#pragma unroll
    for (int ks = 0; ks < 4; ++ks)
        afr[ks] = load_a_frag(lds_chunk, m0, ks * 32, HID, lane);

    const bf16* wqh = wqT + (h << 14);
#pragma unroll
    for (int nt = 0; nt < 8; ++nt) {
        v8f qacc = {};
#pragma unroll
        for (int ks = 0; ks < 4; ++ks)
            qacc = WMMA_BF16(afr[ks], load_b_frag(wqh, nt * 16, ks * 32, HID, lane), qacc);
#pragma unroll
        for (int r = 0; r < 8; ++r)
            lds_kc[(m0 + r + 8 * hi) * HID + nt * 16 + r16] = (bf16)qacc[r];
    }
    v16bf qfrag[4];
#pragma unroll
    for (int ks = 0; ks < 4; ++ks)
        qfrag[ks] = load_a_frag(lds_kc, m0, ks * 32, HID, lane);

    // ---- flash-attention state ----
    float mrow[8], lrow[8];
    v8f o[8];
#pragma unroll
    for (int r = 0; r < 8; ++r) { mrow[r] = -1e30f; lrow[r] = 0.f; }
#pragma unroll
    for (int nt = 0; nt < 8; ++nt) { v8f z = {}; o[nt] = z; }

    const bf16* wkh = wkT + (h << 14);
    const bf16* wvh = wvT + (h << 14);

    for (int c = 0; c < NCHK; ++c) {
        if (c > 0) {
            __syncthreads();
            const float* src = neigh + ((size_t)b * 16 + (c - 1)) * (TIME * HID);
            stage_chunk(lds_chunk, src, tid);
            if (c < NCHK - 1)
                __builtin_prefetch(neigh + ((size_t)b * 16 + c) * (TIME * HID), 0, 1);
            __syncthreads();
#pragma unroll
            for (int ks = 0; ks < 4; ++ks)
                afr[ks] = load_a_frag(lds_chunk, m0, ks * 32, HID, lane);
        }

        // K/V projection of this wave's 16 chunk rows -> Kc (row-major), V^T
#pragma unroll
        for (int nt = 0; nt < 8; ++nt) {
            v8f kacc = {}; v8f vacc = {};
#pragma unroll
            for (int ks = 0; ks < 4; ++ks) {
                v16bf bk = load_b_frag(wkh, nt * 16, ks * 32, HID, lane);
                v16bf bv = load_b_frag(wvh, nt * 16, ks * 32, HID, lane);
                kacc = WMMA_BF16(afr[ks], bk, kacc);
                vacc = WMMA_BF16(afr[ks], bv, vacc);
            }
#pragma unroll
            for (int r = 0; r < 8; ++r) {
                int srow = m0 + r + 8 * hi;
                lds_kc[srow * HID + nt * 16 + r16] = (bf16)kacc[r];
                lds_vt[(nt * 16 + r16) * 64 + srow] = (bf16)vacc[r];
            }
        }
        __syncthreads();

        // S = Q @ Kc^T  (B operand = row-major Kc, n = s', k = e)
        v8f s4[4];
#pragma unroll
        for (int nt = 0; nt < 4; ++nt) {
            v8f sacc = {};
#pragma unroll
            for (int ks = 0; ks < 4; ++ks)
                sacc = WMMA_BF16(qfrag[ks], load_b_frag(lds_kc, nt * 16, ks * 32, HID, lane), sacc);
            s4[nt] = sacc;
        }

        // online softmax update (rows r / r+8 live in lane halves, xor<16 reduces)
#pragma unroll
        for (int r = 0; r < 8; ++r) {
            float cm = -1e30f;
#pragma unroll
            for (int nt = 0; nt < 4; ++nt) {
                float sv = ((mbits >> (nt * 8 + r)) & 1u) ? -1e30f : s4[nt][r];
                s4[nt][r] = sv;
                cm = fmaxf(cm, sv);
            }
#pragma unroll
            for (int off = 1; off < 16; off <<= 1)
                cm = fmaxf(cm, __shfl_xor(cm, off, 32));
            float mn = fmaxf(mrow[r], cm);
            float rescale = __expf(mrow[r] - mn);
            float rs = 0.f;
#pragma unroll
            for (int nt = 0; nt < 4; ++nt) {
                float sv = s4[nt][r];
                float pe = (sv < -1e29f) ? 0.f : __expf(sv - mn);
                rs += pe;
                lds_p[wave][(r + 8 * hi) * 64 + nt * 16 + r16] = (bf16)pe;
            }
#pragma unroll
            for (int off = 1; off < 16; off <<= 1)
                rs += __shfl_xor(rs, off, 32);
            lrow[r] = lrow[r] * rescale + rs;
            mrow[r] = mn;
#pragma unroll
            for (int nt = 0; nt < 8; ++nt) o[nt][r] *= rescale;
        }

        // O += P @ Vc   (A = P via wave-local LDS relayout, B = V^T n-major)
        v16bf pfrag[2];
#pragma unroll
        for (int ks = 0; ks < 2; ++ks)
            pfrag[ks] = load_a_frag(&lds_p[wave][0], 0, ks * 32, 64, lane);
#pragma unroll
        for (int nt = 0; nt < 8; ++nt) {
#pragma unroll
            for (int ks = 0; ks < 2; ++ks)
                o[nt] = WMMA_BF16(pfrag[ks], load_b_frag(lds_vt, nt * 16, ks * 32, 64, lane), o[nt]);
        }
    }

    // ---- finalize O and apply per-head output projection ----
    float invl[8];
#pragma unroll
    for (int r = 0; r < 8; ++r) invl[r] = (lrow[r] > 0.f) ? 1.f / lrow[r] : 0.f;
#pragma unroll
    for (int nt = 0; nt < 8; ++nt)
#pragma unroll
        for (int r = 0; r < 8; ++r) o[nt][r] *= invl[r];

    __syncthreads();                  // done reading kv buffers -> reuse lds_chunk
#pragma unroll
    for (int nt = 0; nt < 8; ++nt)
#pragma unroll
        for (int r = 0; r < 8; ++r)
            lds_chunk[(m0 + r + 8 * hi) * HID + nt * 16 + r16] = (bf16)o[nt][r];

    v16bf ofrag[4];
#pragma unroll
    for (int ks = 0; ks < 4; ++ks)
        ofrag[ks] = load_a_frag(lds_chunk, m0, ks * 32, HID, lane);

#pragma unroll
    for (int nt = 0; nt < 8; ++nt) {
        v8f pacc = {};
#pragma unroll
        for (int ks = 0; ks < 4; ++ks)
            pacc = WMMA_BF16(ofrag[ks],
                             load_b_frag(prjB, nt * 16, h * HID + ks * 32, NHEAD * HID, lane),
                             pacc);
#pragma unroll
        for (int r = 0; r < 8; ++r) {
            int t = m0 + r + 8 * hi;
            atomicAdd(&out[((size_t)b * TIME + t) * HID + nt * 16 + r16], pacc[r]);
        }
    }

    // ---- pass 2: recompute S per chunk, emit A = exp(S-m)/l into Asum ----
    for (int c = 0; c < NCHK; ++c) {
        __syncthreads();
        const float* src = (c == 0) ? node + (size_t)b * (TIME * HID)
                                    : neigh + ((size_t)b * 16 + (c - 1)) * (TIME * HID);
        stage_chunk(lds_chunk, src, tid);
        __syncthreads();
        v16bf a2[4];
#pragma unroll
        for (int ks = 0; ks < 4; ++ks)
            a2[ks] = load_a_frag(lds_chunk, m0, ks * 32, HID, lane);
#pragma unroll
        for (int nt = 0; nt < 8; ++nt) {
            v8f kacc = {};
#pragma unroll
            for (int ks = 0; ks < 4; ++ks)
                kacc = WMMA_BF16(a2[ks], load_b_frag(wkh, nt * 16, ks * 32, HID, lane), kacc);
#pragma unroll
            for (int r = 0; r < 8; ++r)
                lds_kc[(m0 + r + 8 * hi) * HID + nt * 16 + r16] = (bf16)kacc[r];
        }
        __syncthreads();
#pragma unroll
        for (int nt = 0; nt < 4; ++nt) {
            v8f sacc = {};
#pragma unroll
            for (int ks = 0; ks < 4; ++ks)
                sacc = WMMA_BF16(qfrag[ks], load_b_frag(lds_kc, nt * 16, ks * 32, HID, lane), sacc);
#pragma unroll
            for (int r = 0; r < 8; ++r) {
                if ((mbits >> (nt * 8 + r)) & 1u) continue;   // A == 0 there
                float a = __expf(sacc[r] - mrow[r]) * invl[r];
                int t = m0 + r + 8 * hi;
                int s = c * 64 + nt * 16 + r16;
                atomicAdd(&Asum[((size_t)b * TIME + t) * SEQ + s], a);
            }
        }
    }
}

// ---------------------------------------------------------------------------
extern "C" void kernel_launch(void* const* d_in, const int* in_sizes, int n_in,
                              void* d_out, int out_size, void* d_ws, size_t ws_size,
                              hipStream_t stream) {
    (void)in_sizes; (void)n_in; (void)out_size; (void)ws_size;
    const float* node  = (const float*)d_in[0];
    const float* neigh = (const float*)d_in[1];
    const unsigned char* mask = (const unsigned char*)d_in[2];
    const float* w_qs  = (const float*)d_in[3];
    const float* w_ks  = (const float*)d_in[4];
    const float* w_vs  = (const float*)d_in[5];
    const float* prj_w = (const float*)d_in[6];
    const float* prj_b = (const float*)d_in[7];

    float* out  = (float*)d_out;
    float* Asum = out + (size_t)BATCH * TIME * HID;

    // workspace: bf16-transposed weights (1 MB total)
    char* ws = (char*)d_ws;
    bf16* wqT  = (bf16*)(ws);
    bf16* wkT  = (bf16*)(ws + 262144);
    bf16* wvT  = (bf16*)(ws + 524288);
    bf16* prjB = (bf16*)(ws + 786432);

    prep_weights<<<(NHEAD * HID * HID + 255) / 256, 256, 0, stream>>>(
        w_qs, w_ks, w_vs, prj_w, wqT, wkT, wvT, prjB);
    init_out<<<(BATCH * TIME * SEQ + 255) / 256, 256, 0, stream>>>(out, prj_b);

    dim3 grid(BATCH, NHEAD);
    attn_kernel<<<grid, 128, 0, stream>>>(node, neigh, mask,
                                          wqT, wkT, wvT, prjB, out, Asum);
}
